// NystromCausalBasis_18562848653619
// MI455X (gfx1250) — matrix operand order repaired and checked by reading, not compiled, not run on hardware
//
#include <hip/hip_runtime.h>
#include <math.h>

typedef __attribute__((ext_vector_type(16))) _Float16 v16h;
typedef __attribute__((ext_vector_type(8)))  float    v8f;

#define B_ 16
#define T_ 4096
#define D_ 512
#define R_ 256

#define LDS_ROW 80                    // 64B data + 16B pad -> conflict-free b128 reads
#define LDS_BUF (R_ * LDS_ROW)        // 20480 bytes per buffer

// ---- Landmarks: bit-exact replication of np.linspace(0, T-1, R).astype(int64)
struct LMTable { int v[R_]; };
static constexpr LMTable makeLM() {
    LMTable t{};
    double step = 4095.0 / 255.0;
    for (int i = 0; i < R_; ++i) {
        double x = (double)i * step;
        t.v[i] = (int)x;
    }
    t.v[R_ - 1] = 4095;
    return t;
}
__constant__ LMTable cLM = makeLM();

// ------------------------------------------------------------------
// Kernel 1: inverse row norms for all B*T rows (one wave per row)
// ------------------------------------------------------------------
__global__ __launch_bounds__(256) void nrm_kernel(const float* __restrict__ z,
                                                  float* __restrict__ invn) {
    int gid  = blockIdx.x * 256 + threadIdx.x;
    int row  = gid >> 5;
    int lane = threadIdx.x & 31;
    const float4* p = (const float4*)(z + (size_t)row * D_);
    float s = 0.f;
#pragma unroll
    for (int j = 0; j < 4; ++j) {
        float4 v = p[j * 32 + lane];
        s += v.x * v.x + v.y * v.y + v.z * v.z + v.w * v.w;
    }
#pragma unroll
    for (int m = 1; m < 32; m <<= 1) s += __shfl_xor(s, m, 32);
    if (lane == 0) invn[row] = 1.0f / fmaxf(sqrtf(s), 1e-12f);
}

// ------------------------------------------------------------------
// Kernel 2: build normalized landmark matrix Kh (f16), one wave per row
// ------------------------------------------------------------------
struct __align__(8) H4 { _Float16 a, b, c, d; };

__global__ __launch_bounds__(256) void kh_kernel(const float* __restrict__ z,
                                                 _Float16* __restrict__ kh) {
    int gid  = blockIdx.x * 256 + threadIdx.x;
    int w    = gid >> 5;                 // 0 .. B*R-1
    int lane = threadIdx.x & 31;
    int b = w >> 8;
    int r = w & 255;
    int t = cLM.v[r];
    const float4* p = (const float4*)(z + ((size_t)b * T_ + t) * D_);
    float4 v[4];
    float s = 0.f;
#pragma unroll
    for (int j = 0; j < 4; ++j) {
        v[j] = p[j * 32 + lane];
        s += v[j].x * v[j].x + v[j].y * v[j].y + v[j].z * v[j].z + v[j].w * v[j].w;
    }
#pragma unroll
    for (int m = 1; m < 32; m <<= 1) s += __shfl_xor(s, m, 32);
    float inv = 1.0f / fmaxf(sqrtf(s), 1e-12f);
    _Float16* dst = kh + (size_t)w * D_;
#pragma unroll
    for (int j = 0; j < 4; ++j) {
        H4 h;
        h.a = (_Float16)(v[j].x * inv);
        h.b = (_Float16)(v[j].y * inv);
        h.c = (_Float16)(v[j].z * inv);
        h.d = (_Float16)(v[j].w * inv);
        *(H4*)(dst + (size_t)(j * 128 + lane * 4)) = h;
    }
}

// ------------------------------------------------------------------
// Kernel 3: WMMA GEMM with async-to-LDS double-buffered B staging
// Block = 8 waves (one batch), each wave owns a 16-row M strip x 256 cols.
// ------------------------------------------------------------------
__global__ __launch_bounds__(256) void gemm_kernel(const float* __restrict__ z,
                                                   const float* __restrict__ gw,
                                                   const float* __restrict__ ta,
                                                   const _Float16* __restrict__ kh,
                                                   const float* __restrict__ invn,
                                                   float* __restrict__ out) {
    __shared__ __align__(16) char ldsB[2 * LDS_BUF];   // 40 KB

    int tid   = threadIdx.x;
    int lane  = tid & 31;
    int gwave = (blockIdx.x * 256 + tid) >> 5;   // 0 .. 4095
    int b     = gwave >> 8;                      // batch (uniform per block)
    int strip = gwave & 255;
    int t0    = strip * 16;
    int laneN = lane & 15;
    int half  = lane >> 4;

    const float* arow = z + ((size_t)b * T_ + t0 + laneN) * D_;
    float ainv = invn[b * T_ + t0 + laneN];

    // Async prefetch of one 256x32 f16 Kh slice into an LDS buffer.
    // Each of the 256 threads issues 4 async b128 copies (1024 x 16B = 16KB).
    const _Float16* khb = kh + (size_t)b * R_ * D_;
    auto prefetch = [&](int bufi, int kb) {
#pragma unroll
        for (int j = 0; j < 4; ++j) {
            int id = tid + 256 * j;          // chunk id 0..1023
            int r  = id >> 2;                // landmark row
            int c  = id & 3;                 // 16B chunk within 64B row slice
            unsigned int la = (unsigned int)(size_t)
                &ldsB[bufi * LDS_BUF + r * LDS_ROW + c * 16];
            const _Float16* ga = khb + (size_t)r * D_ + kb + c * 8;
            asm volatile("global_load_async_to_lds_b128 %0, %1, off"
                         :: "v"(la), "v"((unsigned long long)(size_t)ga)
                         : "memory");
        }
    };

    v8f acc[16] = {};

    prefetch(0, 0);
    for (int kbi = 0; kbi < 16; ++kbi) {
        int kb = kbi * 32;
        if (kbi < 15) {
            prefetch((kbi + 1) & 1, kb + 32);
            // loads complete in order: <=4 outstanding => current buffer done
            asm volatile("s_wait_asynccnt 4" ::: "memory");
        } else {
            asm volatile("s_wait_asynccnt 0" ::: "memory");
        }
        __syncthreads();   // all waves' slice data visible in LDS

        // A fragment: row m = laneN; lanes 0-15 carry K {0-7,16-23},
        // lanes 16-31 carry K {8-15,24-31} (ISA 16-bit A layout).
        int o0 = kb + 8 * half;
        const float* ap = arow + o0;
        float4 x0 = *(const float4*)(ap);
        float4 x1 = *(const float4*)(ap + 4);
        float4 x2 = *(const float4*)(ap + 16);
        float4 x3 = *(const float4*)(ap + 20);
        v16h a;
        a[0]  = (_Float16)(x0.x * ainv); a[1]  = (_Float16)(x0.y * ainv);
        a[2]  = (_Float16)(x0.z * ainv); a[3]  = (_Float16)(x0.w * ainv);
        a[4]  = (_Float16)(x1.x * ainv); a[5]  = (_Float16)(x1.y * ainv);
        a[6]  = (_Float16)(x1.z * ainv); a[7]  = (_Float16)(x1.w * ainv);
        a[8]  = (_Float16)(x2.x * ainv); a[9]  = (_Float16)(x2.y * ainv);
        a[10] = (_Float16)(x2.z * ainv); a[11] = (_Float16)(x2.w * ainv);
        a[12] = (_Float16)(x3.x * ainv); a[13] = (_Float16)(x3.y * ainv);
        a[14] = (_Float16)(x3.z * ainv); a[15] = (_Float16)(x3.w * ainv);

        const char* bbase = &ldsB[(kbi & 1) * LDS_BUF];
#pragma unroll
        for (int n = 0; n < 16; ++n) {
            // B-matrix (32x16 f16): lane = column n*16+laneN, lanes 0-15 carry
            // K {0-7,16-23}, lanes 16-31 carry K {8-15,24-31} (matches A).
            const uint4* lp = (const uint4*)
                (bbase + (n * 16 + laneN) * LDS_ROW + 16 * half);
            union { uint4 u[2]; v16h v; } bb;
            bb.u[0] = lp[0];
            bb.u[1] = lp[2];
            acc[n] = __builtin_amdgcn_wmma_f32_16x16x32_f16(
                false, a, false, bb.v, (short)0, acc[n], false, false);
        }
        __syncthreads();   // protect buffer the next prefetch overwrites
    }

    // ---- fused epilogue ----
    float g0 = gw[0], g1 = gw[1], g2 = gw[2];
    float gm = fmaxf(g0, fmaxf(g1, g2));
    float e0 = __expf(g0 - gm), e1 = __expf(g1 - gm), e2 = __expf(g2 - gm);
    float es = e0 + e1 + e2;
    float a0 = e0 / es, a1 = e1 / es, a2 = e2 / es;
    float tx = ta[0];
    float a_td = fmaxf(tx, 0.f) + log1pf(__expf(-fabsf(tx)));

    float rowsum[8] = {0.f, 0.f, 0.f, 0.f, 0.f, 0.f, 0.f, 0.f};

#pragma unroll
    for (int n = 0; n < 16; ++n) {
        int rcol = n * 16 + laneN;
        int lmv  = cLM.v[rcol];
        float lmf = (float)lmv;
#pragma unroll
        for (int i = 0; i < 8; ++i) {
            int t = t0 + i + 8 * half;
            float c  = acc[n][i];
            float c2 = c * c;
            float P1 = 0.5f * (1.0f + c);
            float P2 = 0.5f * (3.0f * c2 - 1.0f);
            float P3 = 0.5f * (5.0f * c2 * c - 3.0f * c);
            float phi = a0 * P1 + a1 * P2 + a2 * P3;
            if (lmv < t) {
                float td = fmaxf(((float)t - lmf) * (1.0f / 4096.0f), 0.0f);
                phi *= __expf(-a_td * td);
            } else {
                phi = 0.0f;
            }
            acc[n][i] = phi;
            rowsum[i] += phi;
        }
    }
#pragma unroll
    for (int i = 0; i < 8; ++i) {
#pragma unroll
        for (int m = 1; m < 16; m <<= 1) rowsum[i] += __shfl_xor(rowsum[i], m, 32);
        rowsum[i] = 1.0f / fmaxf(rowsum[i], 1e-6f);
    }

#pragma unroll
    for (int n = 0; n < 16; ++n) {
        int rcol = n * 16 + laneN;
#pragma unroll
        for (int i = 0; i < 8; ++i) {
            int t = t0 + i + 8 * half;
            out[((size_t)(b * T_ + t)) * (R_ + 1) + rcol] = acc[n][i] * rowsum[i];
        }
    }
    if (laneN == 0) {
#pragma unroll
        for (int i = 0; i < 8; ++i) {
            int t = t0 + i + 8 * half;
            out[((size_t)(b * T_ + t)) * (R_ + 1) + R_] = 1.0f;
        }
    }
}

// ------------------------------------------------------------------
extern "C" void kernel_launch(void* const* d_in, const int* in_sizes, int n_in,
                              void* d_out, int out_size, void* d_ws, size_t ws_size,
                              hipStream_t stream) {
    const float* z  = (const float*)d_in[0];
    const float* gw = (const float*)d_in[1];
    const float* ta = (const float*)d_in[2];
    float* out = (float*)d_out;

    // workspace layout: [invn: B*T floats = 256KB][Kh: B*R*D f16 = 4MB]
    float*    invn = (float*)d_ws;
    _Float16* kh   = (_Float16*)((char*)d_ws + (size_t)B_ * T_ * sizeof(float));

    nrm_kernel<<<(B_ * T_) / 8, 256, 0, stream>>>(z, invn);
    kh_kernel<<<(B_ * R_) / 8, 256, 0, stream>>>(z, kh);
    gemm_kernel<<<(B_ * (T_ / 16)) / 8, 256, 0, stream>>>(z, gw, ta, kh, invn, out);
}
